// Net_55980603736397
// MI455X (gfx1250) — compile-verified, matrix operand order checked by reading, not run on hardware
//
#include <hip/hip_runtime.h>
#include <hip/hip_bf16.h>
#include <stddef.h>

// ---------------------------------------------------------------------------
// Problem constants (from reference): B=16, N=2048, K=10, OUT=40
// ---------------------------------------------------------------------------
#define B_  16
#define N_  2048
#define K_  10
#define OUTC 40
#define BNtot (B_ * N_)          // 32768 point rows
#define Me    (BNtot * K_)       // 327680 edge rows

// WMMA tile configuration: 128x64 output tile, 8 waves, 4 wmma / wave / K-step
#define TILE_M 128
#define TILE_N 64
#define TILE_K 32

typedef __attribute__((ext_vector_type(16))) _Float16 v16h;
typedef __attribute__((ext_vector_type(8)))  _Float16 v8h;
typedef __attribute__((ext_vector_type(4)))  _Float16 v4h;
typedef __attribute__((ext_vector_type(8)))  float    v8f;
typedef int gv4i __attribute__((vector_size(16)));   // matches builtin param type

#if __has_builtin(__builtin_amdgcn_global_load_async_to_lds_b128) && \
    __has_builtin(__builtin_amdgcn_s_wait_asynccnt)
#define HAVE_ASYNC_LDS 1
#else
#define HAVE_ASYNC_LDS 0
#endif

// ---------------------------------------------------------------------------
// kNN: temporal dynamic kNN (feature dims d plus the sequence-number coord).
// ---------------------------------------------------------------------------
__global__ void __launch_bounds__(64)
knn_kernel(const float* __restrict__ X, const float* __restrict__ Seq,
           int* __restrict__ Idx, int d, int Npts, int Kn)
{
    extern __shared__ float cxs[];        // [64*d] candidate feats + [64] seq
    float* cx = cxs;
    float* cs = cxs + 64 * d;

    int b = blockIdx.y;
    int i = blockIdx.x * 64 + threadIdx.x;
    const float* qrow = X + ((size_t)b * Npts + i) * d;
    float qseq = Seq[(size_t)b * Npts + i];

    float bd[K_]; int bi[K_];
#pragma unroll
    for (int t = 0; t < K_; ++t) { bd[t] = 3.0e38f; bi[t] = 0; }

    for (int j0 = 0; j0 < Npts; j0 += 64) {
        __syncthreads();
        for (int e = threadIdx.x; e < 64 * d; e += 64)
            cx[e] = X[((size_t)b * Npts + j0) * d + e];
        cs[threadIdx.x] = Seq[(size_t)b * Npts + j0 + threadIdx.x];
        __syncthreads();

        for (int jj = 0; jj < 64; ++jj) {
            int j = j0 + jj;
            if (j == i) continue;                     // no self loops
            const float* cr = cx + jj * d;
            float ds = 0.f;
            for (int c = 0; c < d; ++c) { float t = qrow[c] - cr[c]; ds += t * t; }
            float ts = qseq - cs[jj]; ds += ts * ts;
            if (ds < bd[K_ - 1]) {
                int p = K_ - 1;
                while (p > 0 && bd[p - 1] > ds) { bd[p] = bd[p - 1]; bi[p] = bi[p - 1]; --p; }
                bd[p] = ds; bi[p] = j;
            }
        }
    }
#pragma unroll
    for (int t = 0; t < K_; ++t)
        Idx[((size_t)b * Npts + i) * Kn + t] = bi[t];
}

// ---------------------------------------------------------------------------
// One-time weight prep: W[Kc,Nc] fp32 -> Wt[Nc,Kcp] f16 (transposed, K padded
// to a multiple of 32 with zeros).  Makes the GEMM B-path guard-free and a
// pure byte copy (async-LDS capable).
// ---------------------------------------------------------------------------
__global__ void wconv_kernel(const float* __restrict__ W, _Float16* __restrict__ Wt,
                             int Kc, int Nc, int Kcp)
{
    int i = blockIdx.x * blockDim.x + threadIdx.x;
    if (i >= Nc * Kcp) return;
    int n = i / Kcp, k = i - n * Kcp;
    Wt[i] = (k < Kc) ? (_Float16)W[(size_t)k * Nc + n] : (_Float16)0.f;
}

// ---------------------------------------------------------------------------
// WMMA GEMM:  H[M,Nc] = ReLU(A[M,Kc] @ W[Kc,Nc] + bias)
// mode 0: A is a plain fp32 matrix [M,Kc].
// mode 1: A row r=(b*N+n)*Kn+kk is the edge feature [x_i , x_j - x_i] built
//         on the fly from Xg [B*N, d] and Idx [B*N, Kn]  (Kc == 2*d).
// Wt is the f16-transposed padded weight [Nc][Kcp].
// Fragments are contiguous b128 LDS loads; B tile staged with async-LDS
// copies when the builtin is available.
// ---------------------------------------------------------------------------
__global__ void __launch_bounds__(256)
gemm_wmma_kernel(const float* __restrict__ A,
                 const float* __restrict__ Xg, const int* __restrict__ Idx,
                 const _Float16* __restrict__ Wt, const float* __restrict__ bias,
                 float* __restrict__ Hout,
                 int M, int Kc, int Kcp, int Nc, int mode, int dfeat, int Npts, int Kn)
{
    __shared__ __align__(16) _Float16 As [TILE_M][TILE_K];
    __shared__ __align__(16) _Float16 Wst[TILE_N][TILE_K];

    int tid  = threadIdx.x;
    int wave = tid >> 5;
    int lane = tid & 31;
    int hlf  = lane >> 4;
    int rowBlk = blockIdx.x * TILE_M;
    int colBlk = blockIdx.y * TILE_N;

    v8f acc[4] = {v8f{}, v8f{}, v8f{}, v8f{}};

    int ksteps = Kcp / TILE_K;
    bool fullM = (rowBlk + TILE_M) <= M;

    for (int kt = 0; kt < ksteps; ++kt) {
        int k0 = kt * TILE_K;

        // ---- stage B tile: Wst[col][k] <- Wt[(colBlk+col)*Kcp + k0 + k] ----
        {
            int col = tid >> 2;              // 0..63
            int ck  = (tid & 3) * 8;         // 0,8,16,24
            const _Float16* gsrc = Wt + (size_t)(colBlk + col) * Kcp + k0 + ck;
#if HAVE_ASYNC_LDS
            __builtin_amdgcn_global_load_async_to_lds_b128(
                (__attribute__((address_space(1))) gv4i*)gsrc,
                (__attribute__((address_space(3))) gv4i*)&Wst[col][ck], 0, 0);
#else
            *(v8h*)&Wst[col][ck] = *(const v8h*)gsrc;
#endif
        }

        // ---- stage A tile (128x32 f16, row-major) ----
        bool fastA = fullM && (k0 + TILE_K) <= Kc &&
                     (mode == 0 || (dfeat & 3) == 0);
        if (fastA) {
#pragma unroll
            for (int t = 0; t < 4; ++t) {
                int cid = tid + t * 256;          // 0..1023 float4 chunks
                int r   = cid >> 3;
                int c4  = (cid & 7) * 4;
                int gr  = rowBlk + r;
                int gk  = k0 + c4;
                float4 v;
                if (mode == 0) {
                    v = *(const float4*)(A + (size_t)gr * Kc + gk);
                } else {
                    int bn = gr / Kn;             // b*N + n
                    int kn = gr - bn * Kn;
                    if (gk < dfeat) {
                        v = *(const float4*)(Xg + (size_t)bn * dfeat + gk);
                    } else {
                        int j  = Idx[(size_t)bn * Kn + kn];
                        int bb = bn / Npts;
                        int cc = gk - dfeat;
                        float4 xj = *(const float4*)(Xg + ((size_t)bb * Npts + j) * dfeat + cc);
                        float4 xi = *(const float4*)(Xg + (size_t)bn * dfeat + cc);
                        v = make_float4(xj.x - xi.x, xj.y - xi.y, xj.z - xi.z, xj.w - xi.w);
                    }
                }
                v4h h;
                h[0] = (_Float16)v.x; h[1] = (_Float16)v.y;
                h[2] = (_Float16)v.z; h[3] = (_Float16)v.w;
                *(v4h*)&As[r][c4] = h;
            }
        } else {
            for (int e = tid; e < TILE_M * TILE_K; e += 256) {
                int r  = e >> 5;
                int kk = e & 31;
                int gr = rowBlk + r;
                int gk = k0 + kk;
                float v = 0.f;
                if (gr < M && gk < Kc) {
                    if (mode == 0) {
                        v = A[(size_t)gr * Kc + gk];
                    } else {
                        int bn = gr / Kn;
                        int kn = gr - bn * Kn;
                        if (gk < dfeat) {
                            v = Xg[(size_t)bn * dfeat + gk];
                        } else {
                            int j  = Idx[(size_t)bn * Kn + kn];
                            int bb = bn / Npts;
                            int cc = gk - dfeat;
                            v = Xg[((size_t)bb * Npts + j) * dfeat + cc]
                              - Xg[(size_t)bn * dfeat + cc];
                        }
                    }
                }
                As[r][kk] = (_Float16)v;
            }
        }

#if HAVE_ASYNC_LDS
        __builtin_amdgcn_s_wait_asynccnt(0);
#endif
        __syncthreads();

        // ---- A fragment: two contiguous b128 LDS loads ----
        int arow = wave * 16 + (lane & 15);
        const v8h* ap = (const v8h*)&As[arow][0];
        v16h af = __builtin_shufflevector(ap[hlf], ap[2 + hlf],
                                          0,1,2,3,4,5,6,7,8,9,10,11,12,13,14,15);
        // ---- 4 N sub-tiles ----
#pragma unroll
        for (int sub = 0; sub < 4; ++sub) {
            int col = sub * 16 + (lane & 15);
            const v8h* bp = (const v8h*)&Wst[col][0];
            v16h bf = __builtin_shufflevector(bp[hlf], bp[2 + hlf],
                                              0,1,2,3,4,5,6,7,8,9,10,11,12,13,14,15);
            acc[sub] = __builtin_amdgcn_wmma_f32_16x16x32_f16(
                           false, af, false, bf, (short)0, acc[sub], false, false);
        }
        __syncthreads();
    }

    // ---- epilogue: bias + ReLU (Nc % 64 == 0 -> no column guards) ----
    int nIdx = lane & 15;
#pragma unroll
    for (int sub = 0; sub < 4; ++sub) {
        int gc = colBlk + sub * 16 + nIdx;
        float bv = bias[gc];
#pragma unroll
        for (int r = 0; r < 8; ++r) {
            int grow = rowBlk + wave * 16 + r + hlf * 8;
            if (grow < M) {
                float v = acc[sub][r] + bv;
                Hout[(size_t)grow * Nc + gc] = fmaxf(v, 0.f);
            }
        }
    }
}

// ---------------------------------------------------------------------------
// BatchNorm (training-mode batch stats): one block per channel.
// ---------------------------------------------------------------------------
__global__ void __launch_bounds__(256)
bn_stats_kernel(const float* __restrict__ H, float* __restrict__ meanOut,
                float* __restrict__ varOut, int M, int C)
{
    int c = blockIdx.x, tid = threadIdx.x;
    float s = 0.f, s2 = 0.f;
    for (int r = tid; r < M; r += 256) {
        float v = H[(size_t)r * C + c];
        s += v; s2 += v * v;
    }
    __shared__ float sh[256], sh2[256];
    sh[tid] = s; sh2[tid] = s2;
    __syncthreads();
    for (int off = 128; off > 0; off >>= 1) {
        if (tid < off) { sh[tid] += sh[tid + off]; sh2[tid] += sh2[tid + off]; }
        __syncthreads();
    }
    if (tid == 0) {
        float mu = sh[0] / (float)M;
        meanOut[c] = mu;
        varOut[c]  = sh2[0] / (float)M - mu * mu;
    }
}

__global__ void bn_apply_kernel(float* __restrict__ H,
                                const float* __restrict__ mean,
                                const float* __restrict__ var,
                                const float* __restrict__ gamma,
                                const float* __restrict__ beta,
                                long long total, int C)
{
    long long i = (long long)blockIdx.x * blockDim.x + threadIdx.x;
    if (i >= total) return;
    int c = (int)(i % C);
    float inv = rsqrtf(var[c] + 1e-5f);
    H[i] = gamma[c] * (H[i] - mean[c]) * inv + beta[c];
}

// ---------------------------------------------------------------------------
// Attention logit: L[r] = ReLU( dot(Hin[r,:], w) + b )
// ---------------------------------------------------------------------------
__global__ void att_logit_kernel(const float* __restrict__ Hin,
                                 const float* __restrict__ w, const float* __restrict__ b,
                                 float* __restrict__ L, int M, int d)
{
    int r = blockIdx.x * blockDim.x + threadIdx.x;
    if (r >= M) return;
    float s = b[0];
    for (int c = 0; c < d; ++c) s += Hin[(size_t)r * d + c] * w[c];
    L[r] = fmaxf(s, 0.f);
}

// ---------------------------------------------------------------------------
// softmax over K neighbors, then max_k( alpha_k * m_k )  -> Xo[B*N, d]
// ---------------------------------------------------------------------------
__global__ void att_aggregate_kernel(const float* __restrict__ Mfeat,
                                     const float* __restrict__ L,
                                     float* __restrict__ Xo, int BNrows, int Kn, int d)
{
    long long i = (long long)blockIdx.x * blockDim.x + threadIdx.x;
    if (i >= (long long)BNrows * d) return;
    int bn = (int)(i / d);
    int c  = (int)(i - (long long)bn * d);
    const float* lp = L + (size_t)bn * Kn;
    float mx = -3.0e38f;
    for (int k = 0; k < Kn; ++k) mx = fmaxf(mx, lp[k]);
    float sum = 0.f;
    for (int k = 0; k < Kn; ++k) sum += __expf(lp[k] - mx);
    const float* mp = Mfeat + (size_t)bn * Kn * d + c;
    float best = -3.0e38f;
    for (int k = 0; k < Kn; ++k) {
        float alpha = __expf(lp[k] - mx) / sum;
        best = fmaxf(best, alpha * mp[(size_t)k * d]);
    }
    Xo[i] = best;
}

// ---------------------------------------------------------------------------
// concat [x1|x2|x3] -> [B*N, 448]
// ---------------------------------------------------------------------------
__global__ void concat3_kernel(const float* __restrict__ x1, const float* __restrict__ x2,
                               const float* __restrict__ x3, float* __restrict__ cat,
                               int BNrows)
{
    long long i = (long long)blockIdx.x * blockDim.x + threadIdx.x;
    if (i >= (long long)BNrows * 448) return;
    int bn = (int)(i / 448);
    int c  = (int)(i - (long long)bn * 448);
    float v;
    if (c < 64)       v = x1[(size_t)bn * 64 + c];
    else if (c < 192) v = x2[(size_t)bn * 128 + (c - 64)];
    else              v = x3[(size_t)bn * 256 + (c - 192)];
    cat[i] = v;
}

// ---------------------------------------------------------------------------
// global max pool over N points -> [B, C]
// ---------------------------------------------------------------------------
__global__ void maxpool_kernel(const float* __restrict__ H, float* __restrict__ G,
                               int Bb, int Npts, int C)
{
    int i = blockIdx.x * blockDim.x + threadIdx.x;
    if (i >= Bb * C) return;
    int b = i / C, c = i - b * C;
    float mx = -3.0e38f;
    for (int n = 0; n < Npts; ++n)
        mx = fmaxf(mx, H[((size_t)b * Npts + n) * C + c]);
    G[i] = mx;
}

// ---------------------------------------------------------------------------
// final linear (256->40) + log_softmax per row; 16 rows only.
// ---------------------------------------------------------------------------
__global__ void final_kernel(const float* __restrict__ G, const float* __restrict__ Wf,
                             const float* __restrict__ bf, float* __restrict__ Out, int Mr)
{
    int r = blockIdx.x * blockDim.x + threadIdx.x;
    if (r >= Mr) return;
    float logits[OUTC];
    float mx = -3.0e38f;
    for (int o = 0; o < OUTC; ++o) {
        float s = bf[o];
        for (int c = 0; c < 256; ++c) s += G[(size_t)r * 256 + c] * Wf[(size_t)c * OUTC + o];
        logits[o] = s;
        mx = fmaxf(mx, s);
    }
    float sum = 0.f;
    for (int o = 0; o < OUTC; ++o) sum += __expf(logits[o] - mx);
    float lse = mx + __logf(sum);
    for (int o = 0; o < OUTC; ++o) Out[(size_t)r * OUTC + o] = logits[o] - lse;
}

// ---------------------------------------------------------------------------
// Host orchestration
// ---------------------------------------------------------------------------
extern "C" void kernel_launch(void* const* d_in, const int* in_sizes, int n_in,
                              void* d_out, int out_size, void* d_ws, size_t ws_size,
                              hipStream_t stream)
{
    (void)in_sizes; (void)n_in; (void)out_size; (void)ws_size;
    auto in = [&](int i) -> const float* { return (const float*)d_in[i]; };

    // Input index map (setup_inputs dict insertion order, depth-first):
    // 0 pos, 1 seq
    // conv1_nn L0 w,b,g,bt = 2..5 ; L1 = 6..9 ; L2 = 10..13
    // conv1_att = 14..17
    // conv2_nn  = 18..21 ; conv2_att = 22..25
    // conv3_nn  = 26..29 ; conv3_att = 30..33
    // lin1      = 34..37 ; head1 = 38..41 ; head2 = 42..45
    // final_w = 46 ; final_b = 47
    const float* pos = in(0);
    const float* seq = in(1);

    // ---- workspace carve-out ----
    char* wp = (char*)d_ws;
    auto alloc = [&](size_t bytes) -> void* {
        void* p = (void*)wp;
        wp += (bytes + 255) & ~(size_t)255;
        return p;
    };
    int*   idxb = (int*)  alloc((size_t)Me * 4);
    float* bufA = (float*)alloc((size_t)Me * 256 * 4);
    float* bufB = (float*)alloc((size_t)Me * 256 * 4);
    float* Lbuf = (float*)alloc((size_t)Me * 4);
    float* x1   = (float*)alloc((size_t)BNtot * 64 * 4);
    float* x2   = (float*)alloc((size_t)BNtot * 128 * 4);
    float* x3   = (float*)alloc((size_t)BNtot * 256 * 4);
    float* mean = (float*)alloc(1024 * 4);
    float* var  = (float*)alloc(1024 * 4);
    float* g0   = (float*)alloc(16 * 1024 * 4);
    float* h1   = (float*)alloc(16 * 512 * 4);
    float* h2   = (float*)alloc(16 * 256 * 4);
    float* cat  = bufB;   // [BN,448]  reuses bufB (free at lin1 time)
    float* hlin = bufA;   // [BN,1024] reuses bufA

    // f16 transposed, K-padded weights
    auto Kpad = [](int k) { return (k + 31) & ~31; };
    struct WDesc { int idx, Kc, Nc; };
    const WDesc wd[8] = { {2,6,64}, {6,64,64}, {10,64,64}, {18,128,128},
                          {26,256,256}, {34,448,1024}, {38,1024,512}, {42,512,256} };
    _Float16* wt[8];
    for (int i = 0; i < 8; ++i)
        wt[i] = (_Float16*)alloc((size_t)wd[i].Nc * Kpad(wd[i].Kc) * 2);

    auto eg = [](long long total) { return dim3((unsigned)((total + 255) / 256)); };

    for (int i = 0; i < 8; ++i) {
        int kcp = Kpad(wd[i].Kc);
        long long tot = (long long)wd[i].Nc * kcp;
        wconv_kernel<<<eg(tot), 256, 0, stream>>>(in(wd[i].idx), wt[i],
                                                  wd[i].Kc, wd[i].Nc, kcp);
    }

    auto gemm = [&](const float* Ap, const float* Xg, const int* ip, int widx,
                    const float* bias, float* Hout,
                    int M, int mode, int d) {
        int Kc = wd[widx].Kc, Nc = wd[widx].Nc, Kcp = Kpad(Kc);
        dim3 grid((M + TILE_M - 1) / TILE_M, Nc / TILE_N);
        gemm_wmma_kernel<<<grid, 256, 0, stream>>>(Ap, Xg, ip, wt[widx], bias, Hout,
                                                   M, Kc, Kcp, Nc, mode, d, N_, K_);
    };
    auto bn = [&](float* H, int M, int C, const float* g, const float* bt) {
        bn_stats_kernel<<<C, 256, 0, stream>>>(H, mean, var, M, C);
        long long tot = (long long)M * C;
        bn_apply_kernel<<<eg(tot), 256, 0, stream>>>(H, mean, var, g, bt, tot, C);
    };
    auto knn = [&](const float* X, int d) {
        size_t smem = (size_t)(64 * d + 64) * sizeof(float);
        knn_kernel<<<dim3(N_ / 64, B_), 64, smem, stream>>>(X, seq, idxb, d, N_, K_);
    };
    auto attention = [&](const float* m, int d, int wi, float* xo) {
        att_logit_kernel<<<eg(Me), 256, 0, stream>>>(m, in(wi), in(wi + 1), Lbuf, Me, d);
        bn(Lbuf, Me, 1, in(wi + 2), in(wi + 3));
        att_aggregate_kernel<<<eg((long long)BNtot * d), 256, 0, stream>>>(
            m, Lbuf, xo, BNtot, K_, d);
    };

    // ================= conv1: pos (d=3) -> x1 [BN,64] =================
    knn(pos, 3);
    gemm(nullptr, pos, idxb, 0, in(3), bufA, Me, /*edge*/1, 3);
    bn(bufA, Me, 64, in(4), in(5));
    gemm(bufA, nullptr, nullptr, 1, in(7), bufB, Me, 0, 0);
    bn(bufB, Me, 64, in(8), in(9));
    gemm(bufB, nullptr, nullptr, 2, in(11), bufA, Me, 0, 0);
    bn(bufA, Me, 64, in(12), in(13));
    attention(bufA, 64, 14, x1);

    // ================= conv2: x1 (d=64) -> x2 [BN,128] ================
    knn(x1, 64);
    gemm(nullptr, x1, idxb, 3, in(19), bufA, Me, 1, 64);
    bn(bufA, Me, 128, in(20), in(21));
    attention(bufA, 128, 22, x2);

    // ================= conv3: x2 (d=128) -> x3 [BN,256] ===============
    knn(x2, 128);
    gemm(nullptr, x2, idxb, 4, in(27), bufA, Me, 1, 128);
    bn(bufA, Me, 256, in(28), in(29));
    attention(bufA, 256, 30, x3);

    // ================= lin1 + global max pool =========================
    concat3_kernel<<<eg((long long)BNtot * 448), 256, 0, stream>>>(x1, x2, x3, cat, BNtot);
    gemm(cat, nullptr, nullptr, 5, in(35), hlin, BNtot, 0, 0);
    bn(hlin, BNtot, 1024, in(36), in(37));
    maxpool_kernel<<<eg(B_ * 1024), 256, 0, stream>>>(hlin, g0, B_, N_, 1024);

    // ================= heads (M=16 rows, still WMMA) ==================
    gemm(g0, nullptr, nullptr, 6, in(39), h1, 16, 0, 0);
    bn(h1, 16, 512, in(40), in(41));
    gemm(h1, nullptr, nullptr, 7, in(43), h2, 16, 0, 0);
    bn(h2, 16, 256, in(44), in(45));

    // ================= final linear + log_softmax =====================
    final_kernel<<<1, 32, 0, stream>>>(h2, in(46), in(47), (float*)d_out, B_);
}
